// Model_71880572666542
// MI455X (gfx1250) — matrix operand order repaired and checked by reading, not compiled, not run on hardware
//
#include <hip/hip_runtime.h>

typedef __attribute__((ext_vector_type(16))) _Float16 v16h;
typedef __attribute__((ext_vector_type(8)))  _Float16 v8h;
typedef __attribute__((ext_vector_type(8)))  float    v8f;
typedef __attribute__((ext_vector_type(2)))  float    v2f;

static constexpr int NB = 32, NL = 512, ND = 512, NF = 2048;
static constexpr int NCIN = 21, NCOUT = 21, NTOP = 6;
static constexpr int NM = NB * NL;               // 16384 flattened rows
static constexpr size_t MDSZ = (size_t)NM * ND;  // 8388608 elements per (B,L,D) buffer

// ---------------------------------------------------------------------------
// f32 -> f16 conversion (activation staging + weight conversion)
// ---------------------------------------------------------------------------
__global__ void cvt16_k(const float* __restrict__ x, _Float16* __restrict__ y, int n) {
  int i = blockIdx.x * 256 + threadIdx.x;
  if (i < n) y[i] = (_Float16)x[i];
}

// ---------------------------------------------------------------------------
// f16 WMMA GEMM:  out[m,n] = sum_k A[m,k] * W[n,k]  (+bias[n]) (+res[m,n])
// MODE 0: f32 output (+bias/+residual).  MODE 1: exact-gelu -> f16 output.
// Wave tile: 32(M) x 64(N); block = 4 waves = 128 rows x 64 cols.
// ---------------------------------------------------------------------------
template <int MODE>
__global__ __launch_bounds__(128) void gemm_f16_k(
    const _Float16* __restrict__ A, const _Float16* __restrict__ W,
    const float* __restrict__ bias, const float* res,
    float* outF, _Float16* outH, int N, int Kd) {
  const int wave = threadIdx.x >> 5, lane = threadIdx.x & 31;
  const int laneM = lane & 15, laneHi = lane >> 4;
  const int m0 = blockIdx.y * 128 + wave * 32;
  const int n0 = blockIdx.x * 64;

  const _Float16* rowA[2];
  const _Float16* colW[4];
#pragma unroll
  for (int i = 0; i < 2; ++i)
    rowA[i] = A + (size_t)(m0 + 16 * i + laneM) * Kd + laneHi * 8;
#pragma unroll
  for (int j = 0; j < 4; ++j)
    colW[j] = W + (size_t)(n0 + 16 * j + laneM) * Kd + laneHi * 8;

  v8f acc[2][4] = {};
  union Frag { v16h v; v8h h[2]; };

  for (int k0 = 0; k0 < Kd; k0 += 32) {
    Frag a[2], b[4];
#pragma unroll
    for (int i = 0; i < 2; ++i) {
      a[i].h[0] = *(const v8h*)(rowA[i] + k0);       // lanes0-15: K0..7 ; lanes16-31: K8..15
      a[i].h[1] = *(const v8h*)(rowA[i] + k0 + 16);  // lanes0-15: K16..23 ; lanes16-31: K24..31
    }
#pragma unroll
    for (int j = 0; j < 4; ++j) {
      b[j].h[0] = *(const v8h*)(colW[j] + k0);       // lanes0-15: K0..7 ; lanes16-31: K8..15
      b[j].h[1] = *(const v8h*)(colW[j] + k0 + 16);
    }
#pragma unroll
    for (int i = 0; i < 2; ++i)
#pragma unroll
      for (int j = 0; j < 4; ++j)
        acc[i][j] = __builtin_amdgcn_wmma_f32_16x16x32_f16(
            false, a[i].v, false, b[j].v, (short)0, acc[i][j], false, false);
  }

#pragma unroll
  for (int i = 0; i < 2; ++i) {
#pragma unroll
    for (int j = 0; j < 4; ++j) {
      int n = n0 + 16 * j + laneM;
      float bb = (MODE == 0 && bias) ? bias[n] : 0.0f;
#pragma unroll
      for (int r = 0; r < 8; ++r) {
        int m = m0 + 16 * i + laneHi * 8 + r;  // C layout: VGPR r -> M = r (+8 for hi lanes)
        float v = acc[i][j][r];
        if (MODE == 0) {
          v += bb;
          if (res) v += res[(size_t)m * N + n];
          outF[(size_t)m * N + n] = v;
        } else {
          float g = 0.5f * v * (1.0f + erff(v * 0.70710678118654752f));  // exact gelu
          outH[(size_t)m * N + n] = (_Float16)g;
        }
      }
    }
  }
}

// ---------------------------------------------------------------------------
// f32 WMMA Gram matrix: per batch b, G[b] = Q[b] (512x512) * K[b]^T
// Wave tile 16x64, K-step 4 via v_wmma_f32_16x16x4_f32.
// ---------------------------------------------------------------------------
__global__ __launch_bounds__(128) void gram_f32_k(const float* __restrict__ Q,
                                                  const float* __restrict__ Km,
                                                  float* __restrict__ G) {
  const int wave = threadIdx.x >> 5, lane = threadIdx.x & 31;
  const int laneM = lane & 15, laneHi = lane >> 4;
  const int b = blockIdx.z;
  const int m0 = blockIdx.y * 64 + wave * 16;
  const int n0 = blockIdx.x * 64;
  const float* Qb = Q + ((size_t)b << 18);
  const float* Kb = Km + ((size_t)b << 18);
  float* Gb = G + ((size_t)b << 18);

  const float* rowA = Qb + (size_t)(m0 + laneM) * ND + laneHi * 2;  // A 16x4: lanes0-15 K0,1 ; lanes16-31 K2,3
  const float* colB[4];
#pragma unroll
  for (int j = 0; j < 4; ++j)
    colB[j] = Kb + (size_t)(n0 + 16 * j + laneM) * ND + laneHi * 2;

  v8f acc[4] = {};
#pragma unroll 4
  for (int k0 = 0; k0 < ND; k0 += 4) {
    v2f a = *(const v2f*)(rowA + k0);
#pragma unroll
    for (int j = 0; j < 4; ++j) {
      v2f bb = *(const v2f*)(colB[j] + k0);
      acc[j] = __builtin_amdgcn_wmma_f32_16x16x4_f32(false, a, false, bb, (short)0,
                                                     acc[j], false, false);
    }
  }
#pragma unroll
  for (int j = 0; j < 4; ++j) {
    int n = n0 + 16 * j + laneM;
#pragma unroll
    for (int r = 0; r < 8; ++r) {
      int m = m0 + laneHi * 8 + r;
      Gb[(size_t)m * ND + n] = acc[j][r];
    }
  }
}

// mean_value[b,tau] = (1/512) * sum_s G[b, (s+tau)%512, s]
__global__ __launch_bounds__(256) void diag_mean_k(const float* __restrict__ G,
                                                   float* __restrict__ mv) {
  int b = blockIdx.x >> 9, tau = blockIdx.x & 511;
  const float* Gb = G + ((size_t)b << 18);
  float s = 0.f;
  for (int ss = threadIdx.x; ss < NL; ss += 256) {
    int row = (ss + tau) & 511;
    s += Gb[(size_t)row * NL + ss];
  }
  __shared__ float red[256];
  red[threadIdx.x] = s;
  __syncthreads();
  for (int st = 128; st; st >>= 1) {
    if (threadIdx.x < st) red[threadIdx.x] += red[threadIdx.x + st];
    __syncthreads();
  }
  if (threadIdx.x == 0) mv[blockIdx.x] = red[0] * (1.0f / 512.0f);
}

// top-6 + softmax per batch (serial scan per lane; ties keep earliest index like lax.top_k)
__global__ void topk_k(const float* __restrict__ mv, float* __restrict__ w,
                       int* __restrict__ dly) {
  int b = threadIdx.x;
  if (b >= NB) return;
  const float* m = mv + b * NL;
  float v[NTOP];
  int id[NTOP];
  for (int i = 0; i < NTOP; ++i) { v[i] = -3.4e38f; id[i] = 0; }
  for (int s = 0; s < NL; ++s) {
    float x = m[s];
    if (x > v[NTOP - 1]) {
      int p = NTOP - 1;
      while (p > 0 && x > v[p - 1]) { v[p] = v[p - 1]; id[p] = id[p - 1]; --p; }
      v[p] = x; id[p] = s;
    }
  }
  float mx = v[0], sum = 0.f, e[NTOP];
  for (int i = 0; i < NTOP; ++i) { e[i] = expf(v[i] - mx); sum += e[i]; }
  float inv = 1.f / sum;
  for (int i = 0; i < NTOP; ++i) { w[b * NTOP + i] = e[i] * inv; dly[b * NTOP + i] = id[i]; }
}

// out[b,t,d] = sum_k w[b,k] * V[b, (t+delay[b,k])%512, d]
__global__ void agg_k(const float* __restrict__ V, const float* __restrict__ w,
                      const int* __restrict__ dly, float* __restrict__ out) {
  size_t id = (size_t)blockIdx.x * 256 + threadIdx.x;
  if (id >= MDSZ) return;
  int d = id & 511, t = (id >> 9) & 511, b = (int)(id >> 18);
  const float* Vb = V + ((size_t)b << 18);
  float s = 0.f;
#pragma unroll
  for (int k = 0; k < NTOP; ++k) {
    int tt = (t + dly[b * NTOP + k]) & 511;
    s += w[b * NTOP + k] * Vb[(size_t)tt * ND + d];
  }
  out[id] = s;
}

// series_decomp on (B,512,512): res = x - movavg25(x, replicate-pad); optional trend accumulate
__global__ void decomp_k(const float* __restrict__ x, float* __restrict__ res,
                         float* tacc) {
  size_t id = (size_t)blockIdx.x * 256 + threadIdx.x;
  if (id >= MDSZ) return;
  int d = id & 511, l = (id >> 9) & 511, b = (int)(id >> 18);
  const float* xb = x + ((size_t)b << 18);
  float s = 0.f;
#pragma unroll
  for (int j = -12; j <= 12; ++j) {
    int ll = l + j;
    ll = ll < 0 ? 0 : (ll > 511 ? 511 : ll);
    s += xb[(size_t)ll * ND + d];
  }
  s *= (1.0f / 25.0f);
  res[id] = xb[(size_t)l * ND + d] - s;
  if (tacc) tacc[id] += s;
}

// DataEmbedding_wo_pos: circular conv3 (Cin=21 -> 512) + time linear (4 -> 512)
__global__ void embed_k(const float* __restrict__ x, const float* __restrict__ mark,
                        const float* __restrict__ wconv, const float* __restrict__ wtime,
                        float* __restrict__ out) {
  size_t id = (size_t)blockIdx.x * 256 + threadIdx.x;
  if (id >= MDSZ) return;
  int d = id & 511, l = (id >> 9) & 511, b = (int)(id >> 18);
  const float* wc = wconv + (size_t)d * (NCIN * 3);
  float s = 0.f;
#pragma unroll
  for (int j = 0; j < 3; ++j) {
    int ll = (l + j + 511) & 511;  // x index (l + j - 1) mod L
    const float* xr = x + (size_t)(b * NL + ll) * NCIN;
    for (int c = 0; c < NCIN; ++c) s += xr[c] * wc[c * 3 + j];
  }
  const float* mr = mark + (size_t)(b * NL + l) * 4;
  const float* wt = wtime + (size_t)d * 4;
#pragma unroll
  for (int m2 = 0; m2 < 4; ++m2) s += mr[m2] * wt[m2];
  out[id] = s;
}

// per-channel mean of x_enc over full sequence
__global__ void mean_enc_k(const float* __restrict__ x, float* __restrict__ me) {
  int b = blockIdx.x, c = threadIdx.x;
  if (c >= NCIN) return;
  float s = 0.f;
  for (int l = 0; l < NL; ++l) s += x[(size_t)(b * NL + l) * NCIN + c];
  me[b * NCIN + c] = s * (1.0f / 512.0f);
}

// s_init / t_init: decomp of last 256 rows of x_enc, then [zeros | mean] padding
__global__ void init_dec_k(const float* __restrict__ x, const float* __restrict__ me,
                           float* __restrict__ sinit, float* __restrict__ tinit) {
  size_t id = (size_t)blockIdx.x * 256 + threadIdx.x;
  if (id >= (size_t)NB * NL * NCIN) return;
  int c = (int)(id % NCIN);
  int l = (int)((id / NCIN) % NL);
  int b = (int)(id / ((size_t)NL * NCIN));
  float sv, tv;
  if (l < 256) {
    float s = 0.f;
#pragma unroll
    for (int j = -12; j <= 12; ++j) {
      int ll = l + j;
      ll = ll < 0 ? 0 : (ll > 255 ? 255 : ll);
      s += x[(size_t)(b * NL + 256 + ll) * NCIN + c];
    }
    s *= (1.0f / 25.0f);
    tv = s;
    sv = x[(size_t)(b * NL + 256 + l) * NCIN + c] - s;
  } else {
    sv = 0.f;
    tv = me[b * NCIN + c];
  }
  sinit[id] = sv;
  tinit[id] = tv;
}

// my_Layernorm step 1: standard LN over last dim
__global__ __launch_bounds__(256) void ln_rows_k(const float* __restrict__ x,
                                                 const float* __restrict__ g,
                                                 const float* __restrict__ be,
                                                 float* __restrict__ y) {
  int row = blockIdx.x;
  const float* xr = x + (size_t)row * ND;
  __shared__ float red[256];
  float s = 0.f, s2 = 0.f;
  for (int d = threadIdx.x; d < ND; d += 256) {
    float v = xr[d];
    s += v;
    s2 += v * v;
  }
  red[threadIdx.x] = s;
  __syncthreads();
  for (int st = 128; st; st >>= 1) {
    if (threadIdx.x < st) red[threadIdx.x] += red[threadIdx.x + st];
    __syncthreads();
  }
  float mean = red[0] * (1.0f / 512.0f);
  __syncthreads();
  red[threadIdx.x] = s2;
  __syncthreads();
  for (int st = 128; st; st >>= 1) {
    if (threadIdx.x < st) red[threadIdx.x] += red[threadIdx.x + st];
    __syncthreads();
  }
  float var = red[0] * (1.0f / 512.0f) - mean * mean;
  float rstd = rsqrtf(var + 1e-5f);
  for (int d = threadIdx.x; d < ND; d += 256)
    y[(size_t)row * ND + d] = (xr[d] - mean) * rstd * g[d] + be[d];
}

// my_Layernorm step 2: per-(b,d) mean over sequence, then subtract
__global__ __launch_bounds__(512) void col_mean_k(const float* __restrict__ y,
                                                  float* __restrict__ cm) {
  int b = blockIdx.x, d = threadIdx.x;
  const float* yb = y + ((size_t)b << 18);
  float s = 0.f;
  for (int l = 0; l < NL; ++l) s += yb[(size_t)l * ND + d];
  cm[b * ND + d] = s * (1.0f / 512.0f);
}
__global__ void sub_col_k(float* __restrict__ y, const float* __restrict__ cm) {
  size_t id = (size_t)blockIdx.x * 256 + threadIdx.x;
  if (id >= MDSZ) return;
  int d = id & 511, b = (int)(id >> 18);
  y[id] -= cm[b * ND + d];
}

// trend projection: trend += circ_conv3(TSUM, trend_proj)  (512 -> 21)
__global__ void trend_proj_k(const float* __restrict__ tsum, const float* __restrict__ tp,
                             float* __restrict__ trend) {
  size_t id = (size_t)blockIdx.x * 256 + threadIdx.x;
  if (id >= (size_t)NB * NL * NCOUT) return;
  int c = (int)(id % NCOUT);
  int l = (int)((id / NCOUT) % NL);
  int b = (int)(id / ((size_t)NL * NCOUT));
  float s = 0.f;
#pragma unroll
  for (int j = 0; j < 3; ++j) {
    int ll = (l + j + 511) & 511;
    const float* tr = tsum + (size_t)(b * NL + ll) * ND;
    const float* w = tp + (size_t)c * (ND * 3) + j;
    for (int d = 0; d < ND; ++d) s += tr[d] * w[d * 3];
  }
  trend[id] += s;
}

// final: out = proj(x) + proj_b + trend, sliced to last 256 steps
__global__ void final_k(const float* __restrict__ y, const float* __restrict__ pw,
                        const float* __restrict__ pb, const float* __restrict__ trend,
                        float* __restrict__ out) {
  size_t id = (size_t)blockIdx.x * 256 + threadIdx.x;
  if (id >= (size_t)NB * 256 * NCOUT) return;
  int c = (int)(id % NCOUT);
  int lp = (int)((id / NCOUT) % 256);
  int b = (int)(id / ((size_t)256 * NCOUT));
  int l = 256 + lp;
  const float* yr = y + (size_t)(b * NL + l) * ND;
  const float* w = pw + (size_t)c * ND;
  float s = pb[c] + trend[(size_t)(b * NL + l) * NCOUT + c];
  for (int d = 0; d < ND; ++d) s += yr[d] * w[d];
  out[id] = s;
}

// ---------------------------------------------------------------------------
// host orchestration
// ---------------------------------------------------------------------------
extern "C" void kernel_launch(void* const* d_in, const int* in_sizes, int n_in,
                              void* d_out, int out_size, void* d_ws, size_t ws_size,
                              hipStream_t stream) {
  // ---- detect parameter flattening order ----
  // canonical param list (sorted-pytree order):
  // 0 dec.conv1 1 dec.conv2 2 dc.bk 3 dc.bo 4 dc.bq 5 dc.bv 6 dc.wk 7 dc.wo 8 dc.wq 9 dc.wv
  // 10 ds.bk 11 ds.bo 12 ds.bq 13 ds.bv 14 ds.wk 15 ds.wo 16 ds.wq 17 ds.wv
  // 18 trend_proj 19 dec_norm_b 20 dec_norm_g 21 emb_conv 22 emb_time
  // 23 enc.bk 24 enc.bo 25 enc.bq 26 enc.bv 27 enc.conv1 28 enc.conv2
  // 29 enc.wk 30 enc.wo 31 enc.wq 32 enc.wv 33 enc_norm_b 34 enc_norm_g 35 proj_b 36 proj_w
  int pidx[37];
  int ix_enc = 0, ix_mark_enc = 1, ix_mark_dec = 3;
  if (n_in > 4 && in_sizes[0] == 1048576) {
    // layout B: fully sorted pytree (params first)
    for (int i = 0; i < 37; ++i) pidx[i] = i;
    ix_enc = 38; ix_mark_enc = 40; ix_mark_dec = 39;
  } else if (n_in > 4 && in_sizes[4] == 1048576) {
    // layout A: top-level insertion order, nested sorted
    for (int i = 0; i < 37; ++i) pidx[i] = 4 + i;
  } else {
    // layout C: full insertion order
    const int mapC[37] = {34, 35, 29, 33, 27, 31, 28, 32, 26, 30,
                          21, 25, 19, 23, 20, 24, 18, 22, 36, 38,
                          37, 4, 5, 9, 13, 7, 11, 14, 15, 8,
                          12, 6, 10, 17, 16, 40, 39};
    for (int i = 0; i < 37; ++i) pidx[i] = mapC[i];
  }
  auto PRM = [&](int rel) { return (const float*)d_in[pidx[rel]]; };

  const float* x_enc = (const float*)d_in[ix_enc];
  const float* x_mark_enc = (const float*)d_in[ix_mark_enc];
  const float* x_mark_dec = (const float*)d_in[ix_mark_dec];

  const float* dec_conv1 = PRM(0);  const float* dec_conv2 = PRM(1);
  const float* dc_bk = PRM(2);  const float* dc_bo = PRM(3);
  const float* dc_bq = PRM(4);  const float* dc_bv = PRM(5);
  const float* dc_wk = PRM(6);  const float* dc_wo = PRM(7);
  const float* dc_wq = PRM(8);  const float* dc_wv = PRM(9);
  const float* ds_bk = PRM(10); const float* ds_bo = PRM(11);
  const float* ds_bq = PRM(12); const float* ds_bv = PRM(13);
  const float* ds_wk = PRM(14); const float* ds_wo = PRM(15);
  const float* ds_wq = PRM(16); const float* ds_wv = PRM(17);
  const float* trend_w = PRM(18);
  const float* dec_nb = PRM(19); const float* dec_ng = PRM(20);
  const float* emb_conv = PRM(21); const float* emb_time = PRM(22);
  const float* e_bk = PRM(23); const float* e_bo = PRM(24);
  const float* e_bq = PRM(25); const float* e_bv = PRM(26);
  const float* enc_conv1 = PRM(27); const float* enc_conv2 = PRM(28);
  const float* e_wk = PRM(29); const float* e_wo = PRM(30);
  const float* e_wq = PRM(31); const float* e_wv = PRM(32);
  const float* enc_nb = PRM(33); const float* enc_ng = PRM(34);
  const float* proj_b = PRM(35); const float* proj_w = PRM(36);

  // ---- carve workspace ----
  char* base = (char*)d_ws;
  size_t used = 0;
  auto alloc = [&](size_t bytes) {
    void* r = base + used;
    used += (bytes + 255) & ~(size_t)255;
    return r;
  };
  float* XE = (float*)alloc(MDSZ * 4);
  float* XB = (float*)alloc(MDSZ * 4);
  float* ENCO = (float*)alloc(MDSZ * 4);
  float* Qb = (float*)alloc(MDSZ * 4);
  float* Kb = (float*)alloc(MDSZ * 4);
  float* Vb = (float*)alloc(MDSZ * 4);
  float* CTX = (float*)alloc(MDSZ * 4);  // also holds the Gram matrix G (B*512*512)
  float* TSUM = (float*)alloc(MDSZ * 4);
  _Float16* A16 = (_Float16*)alloc(MDSZ * 2);
  _Float16* MID16 = (_Float16*)alloc((size_t)4096 * NF * 2);
  _Float16* W16 = (_Float16*)alloc((size_t)7340032 * 2);
  float* trend = (float*)alloc((size_t)NB * NL * NCOUT * 4);
  float* sinit = (float*)alloc((size_t)NB * NL * NCOUT * 4);
  float* tinit = (float*)alloc((size_t)NB * NL * NCOUT * 4);
  float* me = (float*)alloc((size_t)NB * NCIN * 4);
  float* mv = (float*)alloc((size_t)NB * NL * 4);
  float* wtop = (float*)alloc((size_t)NB * NTOP * 4);
  int* dtop = (int*)alloc((size_t)NB * NTOP * 4);
  float* cm = (float*)alloc((size_t)NB * ND * 4);
  if (used > ws_size) return;  // insufficient scratch — cannot proceed

  // ---- launch helpers ----
  auto cvt = [&](const float* s, _Float16* d, int n) {
    cvt16_k<<<(n + 255) / 256, 256, 0, stream>>>(s, d, n);
  };
  auto cvtBig = [&](const float* s, _Float16* d) { cvt(s, d, (int)MDSZ); };
  auto gemm = [&](const _Float16* A, const _Float16* W, const float* bias,
                  const float* res, float* out, int M, int N, int Kd) {
    gemm_f16_k<0><<<dim3(N / 64, M / 128), 128, 0, stream>>>(A, W, bias, res, out,
                                                             nullptr, N, Kd);
  };
  auto gemmGelu = [&](const _Float16* A, const _Float16* W, _Float16* out, int M,
                      int N, int Kd) {
    gemm_f16_k<1><<<dim3(N / 64, M / 128), 128, 0, stream>>>(A, W, nullptr, nullptr,
                                                             nullptr, out, N, Kd);
  };
  const int GBIG = (int)((MDSZ + 255) / 256);

  // ---- convert weights to f16 (per-call; ws is volatile) ----
  size_t wo = 0;
  auto wcvt = [&](const float* src, int n) {
    _Float16* dst = W16 + wo;
    cvt(src, dst, n);
    wo += (size_t)n;
    return (const _Float16*)dst;
  };
  const _Float16* e_wq16 = wcvt(e_wq, ND * ND);
  const _Float16* e_wk16 = wcvt(e_wk, ND * ND);
  const _Float16* e_wv16 = wcvt(e_wv, ND * ND);
  const _Float16* e_wo16 = wcvt(e_wo, ND * ND);
  const _Float16* e_c1_16 = wcvt(enc_conv1, NF * ND);
  const _Float16* e_c2_16 = wcvt(enc_conv2, ND * NF);
  const _Float16* s_wq16 = wcvt(ds_wq, ND * ND);
  const _Float16* s_wk16 = wcvt(ds_wk, ND * ND);
  const _Float16* s_wv16 = wcvt(ds_wv, ND * ND);
  const _Float16* s_wo16 = wcvt(ds_wo, ND * ND);
  const _Float16* c_wq16 = wcvt(dc_wq, ND * ND);
  const _Float16* c_wk16 = wcvt(dc_wk, ND * ND);
  const _Float16* c_wv16 = wcvt(dc_wv, ND * ND);
  const _Float16* c_wo16 = wcvt(dc_wo, ND * ND);
  const _Float16* d_c1_16 = wcvt(dec_conv1, NF * ND);
  const _Float16* d_c2_16 = wcvt(dec_conv2, ND * NF);

  // AutoCorrelation layer: x <- x + out_proj(aggregate(topk(corr(Q,K)), V))
  auto attn = [&](float* x, const float* kv, const _Float16* wq, const float* bq,
                  const _Float16* wk, const float* bk, const _Float16* wv,
                  const float* bv, const _Float16* wo16, const float* bo) {
    cvtBig(x, A16);
    gemm(A16, wq, bq, nullptr, Qb, NM, ND, ND);
    if (kv != x) cvtBig(kv, A16);
    gemm(A16, wk, bk, nullptr, Kb, NM, ND, ND);
    gemm(A16, wv, bv, nullptr, Vb, NM, ND, ND);
    gram_f32_k<<<dim3(8, 8, 32), 128, 0, stream>>>(Qb, Kb, CTX);
    diag_mean_k<<<NB * NL, 256, 0, stream>>>(CTX, mv);
    topk_k<<<1, 32, 0, stream>>>(mv, wtop, dtop);
    agg_k<<<GBIG, 256, 0, stream>>>(Vb, wtop, dtop, CTX);
    cvtBig(CTX, A16);
    gemm(A16, wo16, bo, x, x, NM, ND, ND);  // += residual
  };

  // FFN: out = in + conv2(gelu(conv1(in))), chunked over rows to bound MID16
  auto ffn = [&](const float* in, float* out, const _Float16* c1, const _Float16* c2) {
    cvtBig(in, A16);
    for (int ch = 0; ch < 4; ++ch) {
      size_t ro = (size_t)ch * 4096;
      gemmGelu(A16 + ro * ND, c1, MID16, 4096, NF, ND);
      gemm(MID16, c2, nullptr, in + ro * ND, out + ro * ND, 4096, ND, NF);
    }
  };

  // ---- decoder-input prep (depends only on x_enc) ----
  mean_enc_k<<<NB, 32, 0, stream>>>(x_enc, me);
  {
    int n = NB * NL * NCIN;
    init_dec_k<<<(n + 255) / 256, 256, 0, stream>>>(x_enc, me, sinit, tinit);
  }
  hipMemcpyAsync(trend, tinit, (size_t)NB * NL * NCOUT * 4, hipMemcpyDeviceToDevice,
                 stream);

  // ---- encoder (one shared EncoderLayer applied twice) ----
  embed_k<<<GBIG, 256, 0, stream>>>(x_enc, x_mark_enc, emb_conv, emb_time, XE);
  float* cur = XE;
  float* alt = XB;
  for (int it = 0; it < 2; ++it) {
    attn(cur, cur, e_wq16, e_bq, e_wk16, e_bk, e_wv16, e_bv, e_wo16, e_bo);
    decomp_k<<<GBIG, 256, 0, stream>>>(cur, alt, nullptr);
    ffn(alt, cur, e_c1_16, e_c2_16);
    decomp_k<<<GBIG, 256, 0, stream>>>(cur, alt, nullptr);
    float* t = cur; cur = alt; alt = t;
  }
  ln_rows_k<<<NB * NL, 256, 0, stream>>>(cur, enc_ng, enc_nb, ENCO);
  col_mean_k<<<NB, ND, 0, stream>>>(ENCO, cm);
  sub_col_k<<<GBIG, 256, 0, stream>>>(ENCO, cm);

  // ---- decoder (single layer) ----
  embed_k<<<GBIG, 256, 0, stream>>>(sinit, x_mark_dec, emb_conv, emb_time, XE);
  hipMemsetAsync(TSUM, 0, MDSZ * 4, stream);
  attn(XE, XE, s_wq16, ds_bq, s_wk16, ds_bk, s_wv16, ds_bv, s_wo16, ds_bo);
  decomp_k<<<GBIG, 256, 0, stream>>>(XE, XB, TSUM);           // t1
  attn(XB, ENCO, c_wq16, dc_bq, c_wk16, dc_bk, c_wv16, dc_bv, c_wo16, dc_bo);
  decomp_k<<<GBIG, 256, 0, stream>>>(XB, XE, TSUM);           // t2
  ffn(XE, XB, d_c1_16, d_c2_16);
  decomp_k<<<GBIG, 256, 0, stream>>>(XB, XE, TSUM);           // t3
  {
    int n = NB * NL * NCOUT;
    trend_proj_k<<<(n + 255) / 256, 256, 0, stream>>>(TSUM, trend_w, trend);
  }
  ln_rows_k<<<NB * NL, 256, 0, stream>>>(XE, dec_ng, dec_nb, XB);
  col_mean_k<<<NB, ND, 0, stream>>>(XB, cm);
  sub_col_k<<<GBIG, 256, 0, stream>>>(XB, cm);

  {
    int n = NB * 256 * NCOUT;
    final_k<<<(n + 255) / 256, 256, 0, stream>>>(XB, proj_w, proj_b, trend,
                                                 (float*)d_out);
  }
}
